// VQModel_20255065767998
// MI455X (gfx1250) — matrix operand order repaired and checked by reading, not compiled, not run on hardware
//
#include <hip/hip_runtime.h>
#include <hip/hip_bf16.h>

// ---------------------------------------------------------------------------
// VQ video model forward for MI455X (gfx1250), wave32 + WMMA bf16.
// Canonical activation layout everywhere: X[token = t*1024 + s][ch], 64 ch.
// Weights are packed to bf16 once per launch; FFN hidden tensor lives in bf16.
// ---------------------------------------------------------------------------

typedef __attribute__((ext_vector_type(16))) __bf16 v16bf;
typedef __attribute__((ext_vector_type(8)))  float  v8f;

#define NTOK   5120          // 5 * 32 * 32 tokens
#define DMODEL 64
#define NHEAD  4
#define DHEAD  16
#define DFF    2048
#define NLAYER 4
#define NCODE  1024
#define DCODE  16

#if defined(__gfx1250__) && __has_builtin(__builtin_amdgcn_global_load_async_to_lds_b32) && __has_builtin(__builtin_amdgcn_s_wait_asynccnt)
#define HAVE_ASYNC_LDS 1
#else
#define HAVE_ASYNC_LDS 0
#endif

#if HAVE_ASYNC_LDS
typedef __attribute__((address_space(1))) int* gptr_t;
typedef __attribute__((address_space(3))) int* lptr_t;
#endif

__device__ __forceinline__ __bf16 f2bf(float f) {
    union { float f; unsigned u; } a; a.f = f;
    unsigned r = a.u + 0x7FFFu + ((a.u >> 16) & 1u);   // round-to-nearest-even
    union { unsigned short s; __bf16 b; } o; o.s = (unsigned short)(r >> 16);
    return o.b;
}

__device__ __forceinline__ float wave_sum(float v) {   // result in lane 0
    #pragma unroll
    for (int o = 16; o >= 1; o >>= 1) v += __shfl_down(v, o, 32);
    return v;
}
__device__ __forceinline__ float all_sum(float v) {    // result in all lanes
    #pragma unroll
    for (int o = 16; o >= 1; o >>= 1) v += __shfl_xor(v, o, 32);
    return v;
}
__device__ __forceinline__ float all_max(float v) {
    #pragma unroll
    for (int o = 16; o >= 1; o >>= 1) v = fmaxf(v, __shfl_xor(v, o, 32));
    return v;
}

// f32 -> bf16 weight/activation packer
__global__ __launch_bounds__(256)
void k_pack(const float* __restrict__ s, __bf16* __restrict__ d, int n) {
    const int i = blockIdx.x * 256 + threadIdx.x;
    if (i < n) d[i] = f2bf(s[i]);
}

// ---------------------------------------------------------------------------
// GEMM: C[M,N] = A[M,K] @ Wbf[N,K]^T + bias, optional ReLU, optional bf16 out.
// One wave per 16x(NT*16) strip: one A fragment reused across NT B tiles.
// Requires M%16==0, N%(NT*16)==0, K%32==0 (true for all call sites).
// ---------------------------------------------------------------------------
template <typename AT, int NT, bool OUT_BF16>
__global__ __launch_bounds__(32)
void k_gemm(const AT* __restrict__ A, const __bf16* __restrict__ W,
            const float* __restrict__ bias, void* __restrict__ Cv,
            int M, int N, int K, int relu)
{
    const int lane = threadIdx.x;
    const int half = lane >> 4;          // ISA bf16 fragment lane group
    const int lr   = lane & 15;
    const int m0   = blockIdx.y << 4;
    const int n0   = blockIdx.x * (NT * 16);

    v8f acc[NT] = {};
    const AT* arow = A + (size_t)(m0 + lr) * K;

    for (int kb = 0; kb < K; kb += 32) {
        if (kb + 32 < K) __builtin_prefetch(arow + kb + 32, 0, 0);   // global_prefetch_b8
        const int ka = kb + (half << 4);
        // A fragment 16(M)x32(K): lanes 0-15: K[0..15]; lanes 16-31: K[16..31]
        v16bf afrag;
        if constexpr (sizeof(AT) == 4) {
            const float* ap = (const float*)arow;
            #pragma unroll
            for (int e = 0; e < 8; ++e) {
                afrag[e]     = f2bf(ap[ka + e]);
                afrag[e + 8] = f2bf(ap[ka + 8 + e]);
            }
        } else {
            const __bf16* ap = (const __bf16*)arow;
            #pragma unroll
            for (int e = 0; e < 8; ++e) {
                afrag[e]     = ap[ka + e];
                afrag[e + 8] = ap[ka + 8 + e];
            }
        }
        const int kc = kb + (half << 4); // B: lanes 0-15 rows K[0..15], 16-31 K[16..31]
        #pragma unroll
        for (int t = 0; t < NT; ++t) {
            v16bf bfrag = *(const v16bf*)(W + (size_t)(n0 + t * 16 + lr) * K + kc);
            acc[t] = __builtin_amdgcn_wmma_f32_16x16x32_bf16(
                false, afrag, false, bfrag, (short)0, acc[t], false, false);
        }
    }
    // C/D layout: lanes 0-15 rows m0+0..7, lanes 16-31 rows m0+8..15, col = tile*16 + lr
    const int rofs = half << 3;
    #pragma unroll
    for (int t = 0; t < NT; ++t) {
        const int col = n0 + t * 16 + lr;
        const float bv = bias ? bias[col] : 0.0f;
        #pragma unroll
        for (int r = 0; r < 8; ++r) {
            float v = acc[t][r] + bv;
            if (relu) v = fmaxf(v, 0.0f);
            const size_t o = (size_t)(m0 + rofs + r) * N + col;
            if constexpr (OUT_BF16) ((__bf16*)Cv)[o] = f2bf(v);
            else                    ((float*)Cv)[o]  = v;
        }
    }
}

// ---------------------------------------------------------------------------
// Flash-style attention. token(b, j) = b*mult + j*stride.
// 8 waves x 2 queries per block; K/V chunks staged into LDS (async when
// available); per-lane online softmax, one cross-lane combine at the end.
// Reference mask is tril(ones)*-1e4: penalize j <= q when causal.
// ---------------------------------------------------------------------------
#define CHUNK 128
#define QPW   2           // queries per wave
#define QPB   16          // 8 waves * QPW
__global__ __launch_bounds__(256)
void k_attn(const float* __restrict__ qkv, float* __restrict__ out,
            int nseq, int slen, int mult, int stride, int causal)
{
    __shared__ float Ks[CHUNK * 17];     // pad 17 -> conflict-free
    __shared__ float Vs[CHUNK * 17];

    const int b = blockIdx.x / NHEAD, h = blockIdx.x % NHEAD;
    const int tid = threadIdx.x, lane = tid & 31, wid = tid >> 5;
    const int qbase = blockIdx.y * QPB + wid * QPW;
    const float scale = 0.25f;           // 1/sqrt(16)

    float qv[QPW][DHEAD], m[QPW], l[QPW], acc[QPW][DHEAD];
    #pragma unroll
    for (int qi = 0; qi < QPW; ++qi) {
        m[qi] = -3.4e38f; l[qi] = 0.0f;
        #pragma unroll
        for (int d = 0; d < DHEAD; ++d) { acc[qi][d] = 0.0f; qv[qi][d] = 0.0f; }
        const int q = qbase + qi;
        if (q < slen) {
            const float* qp = qkv + (size_t)(b * mult + q * stride) * 192 + h * DHEAD;
            #pragma unroll
            for (int d = 0; d < DHEAD; ++d) qv[qi][d] = qp[d] * scale;
        }
    }

    for (int c0 = 0; c0 < slen; c0 += CHUNK) {
        const int cl = (CHUNK < slen - c0) ? CHUNK : (slen - c0);
        __syncthreads();                                 // previous chunk done
        for (int i = tid; i < cl * DHEAD; i += 256) {
            const int k = i >> 4, d = i & 15;
            const float* kp = qkv + (size_t)(b * mult + (size_t)(c0 + k) * stride) * 192
                            + 64 + h * DHEAD + d;
#if HAVE_ASYNC_LDS
            __builtin_amdgcn_global_load_async_to_lds_b32(
                (gptr_t)kp, (lptr_t)&Ks[k * 17 + d], 0, 0);
            __builtin_amdgcn_global_load_async_to_lds_b32(
                (gptr_t)(kp + 64), (lptr_t)&Vs[k * 17 + d], 0, 0);
#else
            Ks[k * 17 + d] = kp[0];
            Vs[k * 17 + d] = kp[64];
#endif
        }
#if HAVE_ASYNC_LDS
        __builtin_amdgcn_s_wait_asynccnt(0);
#endif
        __syncthreads();

        for (int kk = lane; kk < cl; kk += 32) {
            const int j = c0 + kk;
            #pragma unroll
            for (int qi = 0; qi < QPW; ++qi) {
                const int q = qbase + qi;
                if (q >= slen) continue;
                float s = 0.0f;
                #pragma unroll
                for (int d = 0; d < DHEAD; ++d) s += qv[qi][d] * Ks[kk * 17 + d];
                if (causal && j <= q) s -= 10000.0f;
                const float mn = fmaxf(m[qi], s);
                const float f = __expf(m[qi] - mn), e = __expf(s - mn);
                l[qi] = l[qi] * f + e;
                #pragma unroll
                for (int d = 0; d < DHEAD; ++d)
                    acc[qi][d] = acc[qi][d] * f + e * Vs[kk * 17 + d];
                m[qi] = mn;
            }
        }
    }

    // cross-lane combine and store
    #pragma unroll
    for (int qi = 0; qi < QPW; ++qi) {
        const int q = qbase + qi;
        const float M  = all_max(m[qi]);
        const float f  = __expf(m[qi] - M);
        const float L  = all_sum(l[qi] * f);
        const float iL = 1.0f / L;
        float* op = out + (size_t)(b * mult + q * stride) * DMODEL + h * DHEAD;
        #pragma unroll
        for (int d = 0; d < DHEAD; ++d) {
            const float a = wave_sum(acc[qi][d] * f);
            if (lane == 0 && q < slen) op[d] = a * iL;
        }
    }
}

// residual add + LayerNorm over 64 features: one wave per token, 2 feats/lane
__global__ __launch_bounds__(32)
void k_add_ln(const float* __restrict__ x, const float* __restrict__ res,
              const float* __restrict__ g, const float* __restrict__ b,
              float* __restrict__ out)
{
    const int tok = blockIdx.x, lane = threadIdx.x;
    const int f0 = lane * 2, f1 = f0 + 1;
    float v0 = x[(size_t)tok * 64 + f0], v1 = x[(size_t)tok * 64 + f1];
    if (res) { v0 += res[(size_t)tok * 64 + f0]; v1 += res[(size_t)tok * 64 + f1]; }
    float s = all_sum(v0 + v1);
    const float mean = s * (1.0f / 64.0f);
    const float d0 = v0 - mean, d1 = v1 - mean;
    float vs = all_sum(d0 * d0 + d1 * d1);
    const float inv = rsqrtf(vs * (1.0f / 64.0f) + 1e-5f);
    out[(size_t)tok * 64 + f0] = d0 * inv * g[f0] + b[f0];
    out[(size_t)tok * 64 + f1] = d1 * inv * g[f1] + b[f1];
}

// Conv3d patch embed: x[1,25,3,256,256], w[64,3,5,8,8], strides (5,8,8)
__global__ __launch_bounds__(256)
void k_patch(const float* __restrict__ x, const float* __restrict__ w,
             const float* __restrict__ bias, float* __restrict__ X)
{
    const int idx = blockIdx.x * 256 + threadIdx.x;     // < 5120*64
    const int ch = idx & 63, tok = idx >> 6;
    const int t = tok >> 10, s = tok & 1023, sh = s >> 5, sw = s & 31;
    float acc = bias[ch];
    for (int c = 0; c < 3; ++c)
        for (int kd = 0; kd < 5; ++kd) {
            const float* xp = x + (((size_t)(t * 5 + kd) * 3 + c) * 256 + sh * 8) * 256 + sw * 8;
            const float* wp = w + (((size_t)ch * 3 + c) * 5 + kd) * 64;
            #pragma unroll
            for (int kh = 0; kh < 8; ++kh)
                #pragma unroll
                for (int kw = 0; kw < 8; ++kw)
                    acc += xp[kh * 256 + kw] * wp[kh * 8 + kw];
        }
    X[(size_t)tok * 64 + ch] = acc;
}

// BatchNorm2d (eval) on mapped [NTOK,16]
__global__ __launch_bounds__(256)
void k_bn(float* __restrict__ m, const float* __restrict__ mean,
          const float* __restrict__ var, const float* __restrict__ g,
          const float* __restrict__ b)
{
    const int idx = blockIdx.x * 256 + threadIdx.x;     // < 5120*16
    const int c = idx & 15;
    const float v = (m[idx] - mean[c]) * rsqrtf(var[c] + 1e-5f);
    m[idx] = v * g[c] + b[c];
}

// VQ: one wave per token; lane-parallel codebook scan + shuffle argmin
__global__ __launch_bounds__(32)
void k_vq(const float* __restrict__ M_, const float* __restrict__ cb,
          float* __restrict__ qf, float* __restrict__ commit)
{
    const int tok = blockIdx.x, lane = threadIdx.x;
    float xv[DCODE];
    #pragma unroll
    for (int d = 0; d < DCODE; ++d) xv[d] = M_[(size_t)tok * DCODE + d];
    float best = 3.4e38f; int bidx = NCODE;
    for (int c = lane; c < NCODE; c += 32) {
        const float* cp = cb + (size_t)c * DCODE;
        float d2 = 0.0f;
        #pragma unroll
        for (int d = 0; d < DCODE; ++d) { const float df = xv[d] - cp[d]; d2 += df * df; }
        if (d2 < best) { best = d2; bidx = c; }
    }
    #pragma unroll
    for (int o = 16; o >= 1; o >>= 1) {
        const float ob = __shfl_xor(best, o, 32);
        const int   oi = __shfl_xor(bidx, o, 32);
        if (ob < best || (ob == best && oi < bidx)) { best = ob; bidx = oi; }
    }
    float p = 0.0f;
    if (lane < DCODE) {
        const float q = cb[(size_t)bidx * DCODE + lane];
        qf[(size_t)tok * DCODE + lane] = q;
        const float df = xv[lane] - q; p = df * df;
    }
    p = wave_sum(p);
    if (lane == 0) atomicAdd(commit, p * (1.0f / (5120.0f * 16.0f)));
}

// unmap 16->64 and store in decoder token layout: the reference's raw
// view [5,64,32,32]->[5,1024,64] means dst flat index = t*65536 + o*1024 + hw.
__global__ __launch_bounds__(256)
void k_unmap(const float* __restrict__ qf, const float* __restrict__ w,
             const float* __restrict__ b, float* __restrict__ Y)
{
    const int idx = blockIdx.x * 256 + threadIdx.x;     // < 5120*64
    const int o = idx & 63, tok = idx >> 6;
    float a = b[o];
    #pragma unroll
    for (int c = 0; c < DCODE; ++c) a += qf[(size_t)tok * DCODE + c] * w[o * DCODE + c];
    const int t = tok >> 10, hw = tok & 1023;
    Y[(size_t)t * 65536 + (size_t)o * 1024 + hw] = a;
}

__global__ void k_zero(float* p) { if (threadIdx.x == 0) *p = 0.0f; }

// ---------------------------------------------------------------------------
// Host side
// ---------------------------------------------------------------------------
struct StackP {
    const float *W1, *W2, *Wo, *Wqkv, *bf1, *bf2, *bn1, *bn2, *bo, *bqkv, *g1, *g2, *nb, *ng;
};
struct StackW { const __bf16 *Wqkv, *Wo, *W1, *W2; };

static StackP get_stack(void* const* din, int base) {
    StackP p;
    p.W1   = (const float*)din[base + 0];  p.W2   = (const float*)din[base + 1];
    p.Wo   = (const float*)din[base + 2];  p.Wqkv = (const float*)din[base + 3];
    p.bf1  = (const float*)din[base + 4];  p.bf2  = (const float*)din[base + 5];
    p.bn1  = (const float*)din[base + 6];  p.bn2  = (const float*)din[base + 7];
    p.bo   = (const float*)din[base + 8];  p.bqkv = (const float*)din[base + 9];
    p.g1   = (const float*)din[base + 10]; p.g2   = (const float*)din[base + 11];
    p.nb   = (const float*)din[base + 12]; p.ng   = (const float*)din[base + 13];
    return p;
}

static const __bf16* pack_one(const float* src, int n, __bf16*& dst, hipStream_t s) {
    k_pack<<<(n + 255) / 256, 256, 0, s>>>(src, dst, n);
    const __bf16* r = dst; dst += n; return r;
}
static StackW pack_stack(const StackP& p, __bf16*& dst, hipStream_t s) {
    StackW w;
    w.Wqkv = pack_one(p.Wqkv, NLAYER * 192 * 64, dst, s);
    w.Wo   = pack_one(p.Wo,   NLAYER * 64 * 64,  dst, s);
    w.W1   = pack_one(p.W1,   NLAYER * DFF * 64, dst, s);
    w.W2   = pack_one(p.W2,   NLAYER * 64 * DFF, dst, s);
    return w;
}

static void run_stack(const StackP& p, const StackW& w, float* X, float* qkvB,
                      float* aB, float* oB, __bf16* hB,
                      int nseq, int slen, int mult, int stride, int causal, hipStream_t s)
{
    const int M = NTOK;
    for (int i = 0; i < NLAYER; ++i) {
        k_gemm<float, 4, false><<<dim3(192 / 64, M / 16), 32, 0, s>>>(
            X, w.Wqkv + (size_t)i * 192 * 64, p.bqkv + i * 192, qkvB, M, 192, 64, 0);
        dim3 ga(nseq * NHEAD, (slen + QPB - 1) / QPB);
        k_attn<<<ga, 256, 0, s>>>(qkvB, aB, nseq, slen, mult, stride, causal);
        k_gemm<float, 4, false><<<dim3(1, M / 16), 32, 0, s>>>(
            aB, w.Wo + (size_t)i * 64 * 64, p.bo + i * 64, oB, M, 64, 64, 0);
        k_add_ln<<<M, 32, 0, s>>>(X, oB, p.g1 + i * 64, p.bn1 + i * 64, X);
        k_gemm<float, 4, true><<<dim3(DFF / 64, M / 16), 32, 0, s>>>(
            X, w.W1 + (size_t)i * DFF * 64, p.bf1 + i * DFF, hB, M, DFF, 64, 1);
        k_gemm<__bf16, 4, false><<<dim3(1, M / 16), 32, 0, s>>>(
            hB, w.W2 + (size_t)i * 64 * DFF, p.bf2 + i * 64, oB, M, 64, DFF, 0);
        k_add_ln<<<M, 32, 0, s>>>(X, oB, p.g2 + i * 64, p.bn2 + i * 64, X);
    }
    k_add_ln<<<M, 32, 0, s>>>(X, nullptr, p.ng, p.nb, X);   // final stack LN
}

extern "C" void kernel_launch(void* const* d_in, const int* in_sizes, int n_in,
                              void* d_out, int out_size, void* d_ws, size_t ws_size,
                              hipStream_t stream)
{
    // sorted pytree flatten order of setup_inputs()
    const float* bn_b     = (const float*)d_in[0];
    const float* bn_g     = (const float*)d_in[1];
    const float* bn_mean  = (const float*)d_in[2];
    const float* bn_var   = (const float*)d_in[3];
    const float* codebook = (const float*)d_in[4];
    const StackP dec_sp   = get_stack(d_in, 5);
    const StackP dec_tp   = get_stack(d_in, 19);
    const float* patch_b  = (const float*)d_in[33];
    const float* patch_w  = (const float*)d_in[34];
    const StackP enc_sp   = get_stack(d_in, 35);
    const StackP enc_tp   = get_stack(d_in, 49);
    const float* map_b    = (const float*)d_in[63];
    const float* map_w    = (const float*)d_in[64];
    const float* unmap_b  = (const float*)d_in[65];
    const float* unmap_w  = (const float*)d_in[66];
    const float* x        = (const float*)d_in[67];

    float* ws   = (float*)d_ws;
    float* X    = ws;                           // [5120,64]  f32
    float* QKV  = X   + (size_t)NTOK * 64;      // [5120,192] f32
    float* AB   = QKV + (size_t)NTOK * 192;     // [5120,64]  f32
    float* OB   = AB  + (size_t)NTOK * 64;      // [5120,64]  f32
    float* MB   = OB  + (size_t)NTOK * 64;      // [5120,16]  f32
    float* QF   = MB  + (size_t)NTOK * DCODE;   // [5120,16]  f32
    __bf16* HB  = (__bf16*)(QF + (size_t)NTOK * DCODE);   // [5120,2048] bf16
    __bf16* WPK = HB + (size_t)NTOK * DFF;                // packed weights
    float* out  = (float*)d_out;

    // pack all WMMA weights to bf16 once
    __bf16* wp = WPK;
    const StackW w_enc_sp = pack_stack(enc_sp, wp, stream);
    const StackW w_enc_tp = pack_stack(enc_tp, wp, stream);
    const StackW w_dec_tp = pack_stack(dec_tp, wp, stream);
    const StackW w_dec_sp = pack_stack(dec_sp, wp, stream);
    const __bf16* w_map   = pack_one(map_w, DCODE * 64, wp, stream);

    k_zero<<<1, 1, 0, stream>>>(out + NTOK * 64);   // commit accumulator

    // Encoder
    k_patch<<<(NTOK * 64) / 256, 256, 0, stream>>>(x, patch_w, patch_b, X);
    run_stack(enc_sp, w_enc_sp, X, QKV, AB, OB, HB, 5, 1024, 1024, 1, 0, stream);
    run_stack(enc_tp, w_enc_tp, X, QKV, AB, OB, HB, 1024, 5, 1, 1024, 1, stream);

    // map + BN + VQ + unmap
    k_gemm<float, 1, false><<<dim3(1, NTOK / 16), 32, 0, stream>>>(
        X, w_map, map_b, MB, NTOK, DCODE, 64, 0);
    k_bn<<<(NTOK * DCODE) / 256, 256, 0, stream>>>(MB, bn_mean, bn_var, bn_g, bn_b);
    k_vq<<<NTOK, 32, 0, stream>>>(MB, codebook, QF, out + NTOK * 64);
    k_unmap<<<(NTOK * 64) / 256, 256, 0, stream>>>(QF, unmap_w, unmap_b, X);

    // Decoder (temporal-then-spatial)
    run_stack(dec_tp, w_dec_tp, X, QKV, AB, OB, HB, 1024, 5, 1, 1024, 1, stream);
    run_stack(dec_sp, w_dec_sp, X, QKV, AB, OB, HB, 5, 1024, 1024, 1, 0, stream);

    // final view() is identity on the canonical token layout
    (void)hipMemcpyAsync(out, X, (size_t)NTOK * 64 * sizeof(float),
                         hipMemcpyDeviceToDevice, stream);
}
